// Entity2PredicateCrossAttentionEncoder_88064009437723
// MI455X (gfx1250) — matrix operand order repaired and checked by reading, not compiled, not run on hardware
//
#include <hip/hip_runtime.h>

// ---------------------------------------------------------------------------
// Problem constants (from reference)
// ---------------------------------------------------------------------------
#define BB 64
#define QQ 512
#define VV 80
#define DMODEL 1024
#define NHEAD 8
#define DK 128
#define DINNER 4096
#define ROWS_P (BB * QQ)   // 32768
#define ROWS_E (BB * VV)   // 5120

// ---------------------------------------------------------------------------
// Types / helpers
// ---------------------------------------------------------------------------
typedef __attribute__((ext_vector_type(16))) __bf16 v16bf;
typedef __attribute__((ext_vector_type(8)))  float  v8f;
typedef __attribute__((ext_vector_type(4)))  float  f4;
typedef __attribute__((ext_vector_type(4)))  unsigned int u4;
typedef __attribute__((ext_vector_type(4)))  unsigned short us4;

union FragU { v16bf v; u4 q[2]; };

__device__ inline unsigned short f2bf_rne(float f) {
    unsigned int u = __builtin_bit_cast(unsigned int, f);
    u += 0x7FFFu + ((u >> 16) & 1u);
    return (unsigned short)(u >> 16);
}
__device__ inline float bf2f(unsigned short s) {
    unsigned int u = ((unsigned int)s) << 16;
    return __builtin_bit_cast(float, u);
}
struct BfPair { unsigned short h, l; };
__device__ inline BfPair split2(float v) {
    BfPair p;
    p.h = f2bf_rne(v);
    p.l = f2bf_rne(v - bf2f(p.h));
    return p;
}
__device__ inline f4 pair_f4(us4 h, us4 l) {
    f4 r;
    r.x = bf2f(h.x) + bf2f(l.x);
    r.y = bf2f(h.y) + bf2f(l.y);
    r.z = bf2f(h.z) + bf2f(l.z);
    r.w = bf2f(h.w) + bf2f(l.w);
    return r;
}

// ---------------------------------------------------------------------------
// TDM availability
// ---------------------------------------------------------------------------
#if __has_builtin(__builtin_amdgcn_tensor_load_to_lds)
#define HAVE_TDM 1
#else
#define HAVE_TDM 0
#endif

#if HAVE_TDM
typedef __attribute__((ext_vector_type(4))) unsigned int u32x4;
typedef __attribute__((ext_vector_type(4))) int i32x4;
typedef __attribute__((ext_vector_type(8))) int i32x8;

// Issue one TDM 2-D tile load: 128 rows x 32 bf16 (64B) from a [rows][K] bf16
// tensor starting at (row0, k0), into LDS at lds_off with 16B padding per row
// (LDS row stride = 80B = 40 halfwords). Descriptor per CDNA5 ISA 8.3-8.5.
__device__ inline void tdm_issue(unsigned lds_off, const unsigned short* g,
                                 int row0, int k0, int K, int rows)
{
    unsigned long long ga = (unsigned long long)(const void*)g
                          + ((unsigned long long)(unsigned)row0 * (unsigned)K + (unsigned)k0) * 2ull;
    u32x4 g0;
    g0[0] = 1u;                                                   // count=1, user desc
    g0[1] = lds_off;                                              // lds_addr
    g0[2] = (unsigned)ga;                                         // global_addr[31:0]
    g0[3] = ((unsigned)(ga >> 32) & 0x01FFFFFFu) | 0x80000000u;   // addr[56:32] | type=2
    i32x8 g1;
    // data_size=1 (2B), pad_enable, pad_interval=3 (16 DW), pad_amount=3 (4 DW)
    g1[0] = (int)((1u << 16) | (1u << 20) | (3u << 22) | (3u << 25));
    g1[1] = (int)(((unsigned)K & 0xFFFFu) << 16);                 // tensor_dim0[15:0] @63:48
    g1[2] = (int)((((unsigned)K >> 16) & 0xFFFFu) |               // tensor_dim0[31:16]
                  (((unsigned)rows & 0xFFFFu) << 16));            // tensor_dim1[15:0]
    g1[3] = (int)((((unsigned)rows >> 16) & 0xFFFFu) |            // tensor_dim1[31:16]
                  (32u << 16));                                   // tile_dim0 = 32
    g1[4] = (int)128u;                                            // tile_dim1=128, tile_dim2=0
    g1[5] = (int)(unsigned)K;                                     // tensor_dim0_stride[31:0]
    g1[6] = 0;                                                    // stride hi, dim1_stride lo
    g1[7] = 0;
    i32x4 z4 = (i32x4)0;
#if defined(__clang_major__) && (__clang_major__ >= 23)
    i32x8 z8 = (i32x8)0;
    __builtin_amdgcn_tensor_load_to_lds(g0, g1, z4, z4, z8, 0);
#else
    __builtin_amdgcn_tensor_load_to_lds(g0, g1, z4, z4, 0);
#endif
}
#endif

// ---------------------------------------------------------------------------
// Elementwise f32 -> (bf16 hi, bf16 lo) split, row-major preserved.
// ---------------------------------------------------------------------------
__global__ __launch_bounds__(256)
void split_kernel(const float* __restrict__ in, unsigned short* __restrict__ hi,
                  unsigned short* __restrict__ lo)
{
    const size_t i = ((size_t)blockIdx.x * 256 + threadIdx.x) * 4;
    f4 v = *(const f4*)(in + i);
    us4 h, l;
    BfPair p0 = split2(v.x); h.x = p0.h; l.x = p0.l;
    BfPair p1 = split2(v.y); h.y = p1.h; l.y = p1.l;
    BfPair p2 = split2(v.z); h.z = p2.h; l.z = p2.l;
    BfPair p3 = split2(v.w); h.w = p3.h; l.w = p3.l;
    *(us4*)(hi + i) = h;
    *(us4*)(lo + i) = l;
}

// ---------------------------------------------------------------------------
// Weight transpose + split: W f32 [K][N] -> Thi/Tlo bf16 [N][K]
// ---------------------------------------------------------------------------
__global__ __launch_bounds__(256)
void split_transpose_kernel(const float* __restrict__ W, unsigned short* __restrict__ thi,
                            unsigned short* __restrict__ tlo, int K, int N)
{
    __shared__ float tile[32][33];
    const int k0 = blockIdx.y * 32;
    const int n0 = blockIdx.x * 32;
    const int tx = threadIdx.x & 31;
    const int ty = threadIdx.x >> 5;   // 0..7
#pragma unroll
    for (int j = 0; j < 4; ++j)
        tile[ty + 8*j][tx] = W[(size_t)(k0 + ty + 8*j) * N + n0 + tx];
    __syncthreads();
#pragma unroll
    for (int j = 0; j < 4; ++j) {
        const BfPair p = split2(tile[tx][ty + 8*j]);
        const size_t o = (size_t)(n0 + ty + 8*j) * K + k0 + tx;
        thi[o] = p.h;
        tlo[o] = p.l;
    }
}

// ---------------------------------------------------------------------------
// bf16x3 split GEMM with TDM-staged, double-buffered LDS tiles.
//   out[M,N] = A[M,K] * B^T   with A = Ahi+Alo, B stored [N][K] = Bhi+Blo
// MODE 0: +bias -> bf16 pair out
// MODE 1: +bias +residual -> f32 out
// MODE 2: relu(+bias) -> bf16 pair out
// ---------------------------------------------------------------------------
#define BM 128
#define BN 128
#define BK 32
#define LDT 40                       // LDS row stride (halfwords): 64B data + 16B pad
#define TILE_HW (BM * LDT)           // halfwords per tile buffer
#define TILE_BYTES (TILE_HW * 2)     // 10240 B

template<int MODE>
__global__ __launch_bounds__(256)
void gemm_bf16x3_kernel(const unsigned short* __restrict__ Ahi, const unsigned short* __restrict__ Alo,
                        const unsigned short* __restrict__ Bhi, const unsigned short* __restrict__ Blo,
                        const float* __restrict__ bias, const float* __restrict__ res,
                        float* __restrict__ outF, unsigned short* __restrict__ outHi,
                        unsigned short* __restrict__ outLo, int M, int N, int K)
{
    __shared__ unsigned short smem[2][4][TILE_HW];   // [buf][AsHi,AsLo,BsHi,BsLo]

    const int tid  = threadIdx.x;
    const int lane = tid & 31;
    const int wid  = tid >> 5;
    const int wm   = wid & 1;     // 64-row slab
    const int wn   = wid >> 1;    // 32-col slab
    const int half = lane >> 4;
    const int l15  = lane & 15;
    const int kb   = half * 8;    // A fragment K base (ISA layout)

    const int m0 = blockIdx.y * BM;
    const int n0 = blockIdx.x * BN;
    const int KT = K / BK;

    v8f acc[4][2];
#pragma unroll
    for (int mi = 0; mi < 4; ++mi)
#pragma unroll
        for (int nj = 0; nj < 2; ++nj)
            acc[mi][nj] = (v8f){0.f,0.f,0.f,0.f,0.f,0.f,0.f,0.f};

#if HAVE_TDM
    auto issue = [&](int buf, int kt) {
        if (wid == 0) {
            const unsigned base = (unsigned)(unsigned long long)(const void*)&smem[buf][0][0];
            tdm_issue(base + 0u * TILE_BYTES, Ahi, m0, kt * BK, K, M);
            tdm_issue(base + 1u * TILE_BYTES, Alo, m0, kt * BK, K, M);
            tdm_issue(base + 2u * TILE_BYTES, Bhi, n0, kt * BK, K, N);
            tdm_issue(base + 3u * TILE_BYTES, Blo, n0, kt * BK, K, N);
        }
    };
#else
    auto issue = [&](int buf, int kt) {
        const int row = tid >> 1;
        const int cb  = (tid & 1) * 16;
        const int k0  = kt * BK;
        const size_t ga = (size_t)(m0 + row) * K + k0 + cb;
        const size_t gb = (size_t)(n0 + row) * K + k0 + cb;
        u4* d0 = (u4*)&smem[buf][0][row * LDT + cb];
        u4* d1 = (u4*)&smem[buf][1][row * LDT + cb];
        u4* d2 = (u4*)&smem[buf][2][row * LDT + cb];
        u4* d3 = (u4*)&smem[buf][3][row * LDT + cb];
        d0[0] = *(const u4*)(Ahi + ga); d0[1] = *(const u4*)(Ahi + ga + 8);
        d1[0] = *(const u4*)(Alo + ga); d1[1] = *(const u4*)(Alo + ga + 8);
        d2[0] = *(const u4*)(Bhi + gb); d2[1] = *(const u4*)(Bhi + gb + 8);
        d3[0] = *(const u4*)(Blo + gb); d3[1] = *(const u4*)(Blo + gb + 8);
    };
#endif

    auto compute = [&](int buf) {
        const unsigned short* AsHi = smem[buf][0];
        const unsigned short* AsLo = smem[buf][1];
        const unsigned short* BsHi = smem[buf][2];
        const unsigned short* BsLo = smem[buf][3];
        FragU ahi[4], alo[4];
#pragma unroll
        for (int mi = 0; mi < 4; ++mi) {
            const int base = (wm*64 + mi*16 + l15) * LDT + kb;
            ahi[mi].q[0] = *(const u4*)&AsHi[base];
            ahi[mi].q[1] = *(const u4*)&AsHi[base + 16];
            alo[mi].q[0] = *(const u4*)&AsLo[base];
            alo[mi].q[1] = *(const u4*)&AsLo[base + 16];
        }
#pragma unroll
        for (int nj = 0; nj < 2; ++nj) {
            const int base = (wn*32 + nj*16 + l15) * LDT + kb;
            FragU bhi, blo;
            bhi.q[0] = *(const u4*)&BsHi[base];
            bhi.q[1] = *(const u4*)&BsHi[base + 16];
            blo.q[0] = *(const u4*)&BsLo[base];
            blo.q[1] = *(const u4*)&BsLo[base + 16];
#pragma unroll
            for (int mi = 0; mi < 4; ++mi) {
                acc[mi][nj] = __builtin_amdgcn_wmma_f32_16x16x32_bf16(
                    false, alo[mi].v, false, bhi.v, (short)0, acc[mi][nj], false, false);
                acc[mi][nj] = __builtin_amdgcn_wmma_f32_16x16x32_bf16(
                    false, ahi[mi].v, false, blo.v, (short)0, acc[mi][nj], false, false);
                acc[mi][nj] = __builtin_amdgcn_wmma_f32_16x16x32_bf16(
                    false, ahi[mi].v, false, bhi.v, (short)0, acc[mi][nj], false, false);
            }
        }
    };

    // pipeline: prefetch next tile while computing current
    issue(0, 0);
#if HAVE_TDM
    if (wid == 0) __builtin_amdgcn_s_wait_tensorcnt(0);
#endif
    __syncthreads();
    for (int kt = 0; kt < KT; ++kt) {
        if (kt + 1 < KT) issue((kt + 1) & 1, kt + 1);
        compute(kt & 1);
#if HAVE_TDM
        if (wid == 0 && kt + 1 < KT) __builtin_amdgcn_s_wait_tensorcnt(0);
#endif
        __syncthreads();
    }

    // epilogue: C/D layout m = i + 8*half, n = l15
#pragma unroll
    for (int mi = 0; mi < 4; ++mi) {
#pragma unroll
        for (int nj = 0; nj < 2; ++nj) {
            const int col = n0 + wn*32 + nj*16 + l15;
            const float bv = bias[col];
#pragma unroll
            for (int i = 0; i < 8; ++i) {
                const int row = m0 + wm*64 + mi*16 + half*8 + i;
                const size_t o = (size_t)row * N + col;
                float v = acc[mi][nj][i] + bv;
                if (MODE == 1) {
                    outF[o] = v + res[o];
                } else {
                    if (MODE == 2) v = fmaxf(v, 0.0f);
                    const BfPair p = split2(v);
                    outHi[o] = p.h;
                    outLo[o] = p.l;
                }
            }
        }
    }
}

// ---------------------------------------------------------------------------
// Sparse pair attention: each query attends only to its 2 pair entities.
// q/k/v are bf16 hi/lo pairs; ctx written as bf16 pair (may alias q buffers).
// ---------------------------------------------------------------------------
__global__ __launch_bounds__(256)
void attn_gather_kernel(const unsigned short* __restrict__ qhi, const unsigned short* __restrict__ qlo,
                        const unsigned short* __restrict__ khi, const unsigned short* __restrict__ klo,
                        const unsigned short* __restrict__ vhi, const unsigned short* __restrict__ vlo,
                        const int* __restrict__ rel,
                        unsigned short* __restrict__ chi, unsigned short* __restrict__ clo)
{
    const int bq   = blockIdx.x;          // 0 .. B*Q-1
    const int b    = bq >> 9;             // Q = 512
    const int h    = threadIdx.x >> 5;    // head = wave id
    const int lane = threadIdx.x & 31;

    const int i0 = rel[bq * 2 + 0];
    const int i1 = rel[bq * 2 + 1];

    const size_t qoff  = (size_t)bq * DMODEL + h * DK + lane * 4;
    const size_t e0off = (size_t)(b * VV + i0) * DMODEL + h * DK + lane * 4;
    const size_t e1off = (size_t)(b * VV + i1) * DMODEL + h * DK + lane * 4;

    f4 q  = pair_f4(*(const us4*)(qhi + qoff),  *(const us4*)(qlo + qoff));
    f4 k0 = pair_f4(*(const us4*)(khi + e0off), *(const us4*)(klo + e0off));
    f4 k1 = pair_f4(*(const us4*)(khi + e1off), *(const us4*)(klo + e1off));

    float p0 = q.x*k0.x + q.y*k0.y + q.z*k0.z + q.w*k0.w;
    float p1 = q.x*k1.x + q.y*k1.y + q.z*k1.z + q.w*k1.w;
#pragma unroll
    for (int off = 16; off >= 1; off >>= 1) {
        p0 += __shfl_xor(p0, off, 32);
        p1 += __shfl_xor(p1, off, 32);
    }
    const float scale = 0.08838834764831845f;   // 1/sqrt(128)
    const float s0 = p0 * scale, s1 = p1 * scale;
    const float m  = fmaxf(s0, s1);
    const float e0 = __expf(s0 - m), e1 = __expf(s1 - m);
    const float inv = 1.0f / (e0 + e1);
    const float w0 = e0 * inv, w1 = e1 * inv;

    f4 v0 = pair_f4(*(const us4*)(vhi + e0off), *(const us4*)(vlo + e0off));
    f4 v1 = pair_f4(*(const us4*)(vhi + e1off), *(const us4*)(vlo + e1off));
    us4 oh, ol;
    BfPair c0 = split2(w0*v0.x + w1*v1.x); oh.x = c0.h; ol.x = c0.l;
    BfPair c1 = split2(w0*v0.y + w1*v1.y); oh.y = c1.h; ol.y = c1.l;
    BfPair c2 = split2(w0*v0.z + w1*v1.z); oh.z = c2.h; ol.z = c2.l;
    BfPair c3 = split2(w0*v0.w + w1*v1.w); oh.w = c3.h; ol.w = c3.l;
    *(us4*)(chi + qoff) = oh;
    *(us4*)(clo + qoff) = ol;
}

// ---------------------------------------------------------------------------
// LayerNorm over D_MODEL=1024; in-place f32, optionally also bf16 pair out.
// ---------------------------------------------------------------------------
template<bool WP>
__global__ __launch_bounds__(256)
void layernorm_kernel(float* __restrict__ x, const float* __restrict__ g,
                      const float* __restrict__ bta,
                      unsigned short* __restrict__ ohi, unsigned short* __restrict__ olo)
{
    const int row = blockIdx.x;
    const int tid = threadIdx.x;
    float* xr = x + (size_t)row * DMODEL;

    f4 v = *(const f4*)(xr + tid * 4);
    float s  = v.x + v.y + v.z + v.w;
    float ss = v.x*v.x + v.y*v.y + v.z*v.z + v.w*v.w;
#pragma unroll
    for (int off = 16; off >= 1; off >>= 1) {
        s  += __shfl_xor(s,  off, 32);
        ss += __shfl_xor(ss, off, 32);
    }
    __shared__ float sb[8], sb2[8];
    if ((tid & 31) == 0) { sb[tid >> 5] = s; sb2[tid >> 5] = ss; }
    __syncthreads();
    float ts = 0.f, tss = 0.f;
#pragma unroll
    for (int i = 0; i < 8; ++i) { ts += sb[i]; tss += sb2[i]; }

    const float mean = ts * (1.0f / DMODEL);
    const float var  = tss * (1.0f / DMODEL) - mean * mean;
    const float rs   = rsqrtf(var + 1e-5f);

    f4 gg = *(const f4*)(g + tid * 4);
    f4 bb = *(const f4*)(bta + tid * 4);
    f4 o;
    o.x = (v.x - mean) * rs * gg.x + bb.x;
    o.y = (v.y - mean) * rs * gg.y + bb.y;
    o.z = (v.z - mean) * rs * gg.z + bb.z;
    o.w = (v.w - mean) * rs * gg.w + bb.w;
    *(f4*)(xr + tid * 4) = o;
    if (WP) {
        const size_t base = (size_t)row * DMODEL + tid * 4;
        us4 h, l;
        BfPair p0 = split2(o.x); h.x = p0.h; l.x = p0.l;
        BfPair p1 = split2(o.y); h.y = p1.h; l.y = p1.l;
        BfPair p2 = split2(o.z); h.z = p2.h; l.z = p2.l;
        BfPair p3 = split2(o.w); h.w = p3.h; l.w = p3.l;
        *(us4*)(ohi + base) = h;
        *(us4*)(olo + base) = l;
    }
}

// ---------------------------------------------------------------------------
// Launch pipeline
// ---------------------------------------------------------------------------
extern "C" void kernel_launch(void* const* d_in, const int* in_sizes, int n_in,
                              void* d_out, int out_size, void* d_ws, size_t ws_size,
                              hipStream_t stream)
{
    (void)in_sizes; (void)n_in; (void)out_size; (void)ws_size;

    const float* pred  = (const float*)d_in[0];
    const float* ent   = (const float*)d_in[1];
    const int*   rel   = (const int*)  d_in[2];
    const float* w_q   = (const float*)d_in[3];
    const float* b_q   = (const float*)d_in[4];
    const float* w_k   = (const float*)d_in[5];
    const float* b_k   = (const float*)d_in[6];
    const float* w_v   = (const float*)d_in[7];
    const float* b_v   = (const float*)d_in[8];
    const float* w_o   = (const float*)d_in[9];
    const float* b_o   = (const float*)d_in[10];
    const float* ln1_g = (const float*)d_in[11];
    const float* ln1_b = (const float*)d_in[12];
    const float* w1    = (const float*)d_in[13];
    const float* b1    = (const float*)d_in[14];
    const float* w2    = (const float*)d_in[15];
    const float* b2    = (const float*)d_in[16];
    const float* ln2_g = (const float*)d_in[17];
    const float* ln2_b = (const float*)d_in[18];
    float* out = (float*)d_out;

    // ---- workspace carve-out ----
    char* ws = (char*)d_ws;
    size_t cur = 0;
    auto alloc = [&](size_t bytes) -> char* {
        char* p = ws + cur;
        cur += (bytes + 255) & ~(size_t)255;
        return p;
    };
    typedef unsigned short u16;
    const size_t NP = (size_t)ROWS_P * DMODEL;    // 33.5M
    const size_t NE = (size_t)ROWS_E * DMODEL;    // 5.2M
    const size_t NH = (size_t)ROWS_P * DINNER;    // 134M

    u16* predhi = (u16*)alloc(NP * 2);  u16* predlo = (u16*)alloc(NP * 2);
    u16* enthi  = (u16*)alloc(NE * 2);  u16* entlo  = (u16*)alloc(NE * 2);
    u16* qhi    = (u16*)alloc(NP * 2);  u16* qlo    = (u16*)alloc(NP * 2);   // reused as ctx
    u16* khi    = (u16*)alloc(NE * 2);  u16* klo    = (u16*)alloc(NE * 2);
    u16* vhi    = (u16*)alloc(NE * 2);  u16* vlo    = (u16*)alloc(NE * 2);
    float* xbuf = (float*)alloc(NP * 4);
    u16* xhi    = (u16*)alloc(NP * 2);  u16* xlo    = (u16*)alloc(NP * 2);
    u16* hhi    = (u16*)alloc(NH * 2);  u16* hlo    = (u16*)alloc(NH * 2);
    u16* wqThi  = (u16*)alloc((size_t)DMODEL*DMODEL*2); u16* wqTlo = (u16*)alloc((size_t)DMODEL*DMODEL*2);
    u16* wkThi  = (u16*)alloc((size_t)DMODEL*DMODEL*2); u16* wkTlo = (u16*)alloc((size_t)DMODEL*DMODEL*2);
    u16* wvThi  = (u16*)alloc((size_t)DMODEL*DMODEL*2); u16* wvTlo = (u16*)alloc((size_t)DMODEL*DMODEL*2);
    u16* woThi  = (u16*)alloc((size_t)DMODEL*DMODEL*2); u16* woTlo = (u16*)alloc((size_t)DMODEL*DMODEL*2);
    u16* w1Thi  = (u16*)alloc((size_t)DMODEL*DINNER*2); u16* w1Tlo = (u16*)alloc((size_t)DMODEL*DINNER*2);
    u16* w2Thi  = (u16*)alloc((size_t)DINNER*DMODEL*2); u16* w2Tlo = (u16*)alloc((size_t)DINNER*DMODEL*2);

    const dim3 blk(256);

    // ---- one-time operand splits ----
    split_kernel<<<dim3((unsigned)(NP / 1024)), blk, 0, stream>>>(pred, predhi, predlo);
    split_kernel<<<dim3((unsigned)(NE / 1024)), blk, 0, stream>>>(ent, enthi, entlo);
    split_transpose_kernel<<<dim3(DMODEL/32, DMODEL/32), blk, 0, stream>>>(w_q, wqThi, wqTlo, DMODEL, DMODEL);
    split_transpose_kernel<<<dim3(DMODEL/32, DMODEL/32), blk, 0, stream>>>(w_k, wkThi, wkTlo, DMODEL, DMODEL);
    split_transpose_kernel<<<dim3(DMODEL/32, DMODEL/32), blk, 0, stream>>>(w_v, wvThi, wvTlo, DMODEL, DMODEL);
    split_transpose_kernel<<<dim3(DMODEL/32, DMODEL/32), blk, 0, stream>>>(w_o, woThi, woTlo, DMODEL, DMODEL);
    split_transpose_kernel<<<dim3(DINNER/32, DMODEL/32), blk, 0, stream>>>(w1, w1Thi, w1Tlo, DMODEL, DINNER);
    split_transpose_kernel<<<dim3(DMODEL/32, DINNER/32), blk, 0, stream>>>(w2, w2Thi, w2Tlo, DINNER, DMODEL);

    // ---- Q/K/V projections -> bf16 pairs ----
    gemm_bf16x3_kernel<0><<<dim3(DMODEL/BN, ROWS_P/BM), blk, 0, stream>>>(
        predhi, predlo, wqThi, wqTlo, b_q, nullptr, nullptr, qhi, qlo, ROWS_P, DMODEL, DMODEL);
    gemm_bf16x3_kernel<0><<<dim3(DMODEL/BN, ROWS_E/BM), blk, 0, stream>>>(
        enthi, entlo, wkThi, wkTlo, b_k, nullptr, nullptr, khi, klo, ROWS_E, DMODEL, DMODEL);
    gemm_bf16x3_kernel<0><<<dim3(DMODEL/BN, ROWS_E/BM), blk, 0, stream>>>(
        enthi, entlo, wvThi, wvTlo, b_v, nullptr, nullptr, vhi, vlo, ROWS_E, DMODEL, DMODEL);

    // ---- masked attention = gather + 2-way softmax; ctx overwrites q pair ----
    attn_gather_kernel<<<dim3(ROWS_P), blk, 0, stream>>>(qhi, qlo, khi, klo, vhi, vlo, rel, qhi, qlo);

    // ---- output projection + residual(pred) -> xbuf f32 ; LN1 (also emits pair) ----
    gemm_bf16x3_kernel<1><<<dim3(DMODEL/BN, ROWS_P/BM), blk, 0, stream>>>(
        qhi, qlo, woThi, woTlo, b_o, pred, xbuf, nullptr, nullptr, ROWS_P, DMODEL, DMODEL);
    layernorm_kernel<true><<<dim3(ROWS_P), blk, 0, stream>>>(xbuf, ln1_g, ln1_b, xhi, xlo);

    // ---- FFN1 with ReLU -> h pair ----
    gemm_bf16x3_kernel<2><<<dim3(DINNER/BN, ROWS_P/BM), blk, 0, stream>>>(
        xhi, xlo, w1Thi, w1Tlo, b1, nullptr, nullptr, hhi, hlo, ROWS_P, DINNER, DMODEL);

    // ---- FFN2 + residual(x) -> out f32 ; LN2 in place ----
    gemm_bf16x3_kernel<1><<<dim3(DMODEL/BN, ROWS_P/BM), blk, 0, stream>>>(
        hhi, hlo, w2Thi, w2Tlo, b2, xbuf, out, nullptr, nullptr, ROWS_P, DMODEL, DINNER);
    layernorm_kernel<false><<<dim3(ROWS_P), blk, 0, stream>>>(out, ln2_g, ln2_b, nullptr, nullptr);
}
